// BasicGCN_33277406610019
// MI455X (gfx1250) — matrix operand order repaired and checked by reading, not compiled, not run on hardware
//
#include <hip/hip_runtime.h>
#include <hip/hip_bf16.h>

typedef __attribute__((ext_vector_type(16))) _Float16 v16h;
typedef __attribute__((ext_vector_type(8)))  _Float16 v8h;
typedef __attribute__((ext_vector_type(8)))  float    v8f;

#define HID  128
#define LDA  136   // halves; A-tile row stride in LDS (272B rows -> 16B aligned)
#define LDBT 144   // halves; W^T row stride in LDS (288B rows -> 32B aligned)

// ---------------------------------------------------------------------------
// C[M x 128] = relu?( A[M x 128] * W[128 x 128] + bias )  via v_wmma f16
// block = 128 threads (4 waves). Block tile: 64 rows x 128 cols.
// Wave w owns rows [64*bx + 16w, +16), all 8 column tiles (8 acc of v8f).
// ---------------------------------------------------------------------------
__global__ __launch_bounds__(128)
void gemm128_wmma_kernel(const float* __restrict__ A, const float* __restrict__ W,
                         const float* __restrict__ bias, float* __restrict__ C,
                         int M, int relu)
{
    __shared__ alignas(64) _Float16 ldsA [64  * LDA ];
    __shared__ alignas(64) _Float16 ldsBt[128 * LDBT];

    const int tid  = threadIdx.x;
    const int wave = tid >> 5;
    const int lane = tid & 31;
    const int kh   = lane >> 4;   // K-half selector (ISA 16-bit A/B lane layout)
    const int mn   = lane & 15;   // M for A-frag, N for B-frag, col for C
    const int rowBlock = blockIdx.x * 64;

    // Stage W transposed as f16: ldsBt[n][k] = W[k][n]  (reads coalesced over n)
    for (int i = tid; i < 128 * 128; i += 128) {
        int k = i >> 7, n = i & 127;
        ldsBt[n * LDBT + k] = (_Float16)W[k * 128 + n];
    }
    // Stage A tile (64 rows x 128 K) as f16, zero-pad past M
    for (int i = tid; i < 64 * 128; i += 128) {
        int r = i >> 7, c = i & 127;
        int row = rowBlock + r;
        float v = (row < M) ? A[(size_t)row * 128 + c] : 0.0f;
        ldsA[r * LDA + c] = (_Float16)v;
    }
    __syncthreads();

    v8f acc[8] = {};

#pragma unroll
    for (int kt = 0; kt < 4; ++kt) {                  // K = 128 -> 4 x K32
        // A fragment 16x32: lane(mn,kh): elems 0..7 -> K = 32kt + 8kh + e,
        //                               elems 8..15 -> K = 32kt + 16 + 8kh + e
        const _Float16* ap = &ldsA[(wave * 16 + mn) * LDA + kt * 32 + kh * 8];
        v8h alo = *(const v8h*)(ap);
        v8h ahi = *(const v8h*)(ap + 16);
        v16h a  = __builtin_shufflevector(alo, ahi,
                     0,1,2,3,4,5,6,7,8,9,10,11,12,13,14,15);
#pragma unroll
        for (int nt = 0; nt < 8; ++nt) {
            // B fragment 32x16 (from W^T): lane(mn,kh): elems 0..7 -> K = 32kt + 8kh + e, elems 8..15 -> K = 32kt + 16 + 8kh + e
            const _Float16* bp = &ldsBt[(nt * 16 + mn) * LDBT + kt * 32 + kh * 8];
            v8h blo = *(const v8h*)(bp);
            v8h bhi = *(const v8h*)(bp + 16);
            v16h b = __builtin_shufflevector(blo, bhi,
                         0,1,2,3,4,5,6,7,8,9,10,11,12,13,14,15);
            acc[nt] = __builtin_amdgcn_wmma_f32_16x16x32_f16(
                          false, a, false, b, (short)0, acc[nt], false, false);
        }
    }

    // C/D layout: VGPR r of lane(mn,kh) = element (row = 8*kh + r, col = mn)
#pragma unroll
    for (int nt = 0; nt < 8; ++nt) {
        const int ccol = nt * 16 + mn;
        const float badd = bias ? bias[ccol] : 0.0f;
#pragma unroll
        for (int r = 0; r < 8; ++r) {
            int row = rowBlock + wave * 16 + kh * 8 + r;
            if (row < M) {
                float v = acc[nt][r] + badd;
                if (relu) v = fmaxf(v, 0.0f);
                C[(size_t)row * 128 + ccol] = v;
            }
        }
    }
}

// ---------------------------------------------------------------------------
// Degree / norm kernels
// ---------------------------------------------------------------------------
__global__ void deg_init_kernel(float* deg, int n) {
    int i = blockIdx.x * blockDim.x + threadIdx.x;
    if (i < n) deg[i] = 1.0f;                       // self-loop
}
__global__ void deg_count_kernel(const int* __restrict__ col, float* deg, int E) {
    int e = blockIdx.x * blockDim.x + threadIdx.x;
    if (e < E) atomicAdd(&deg[col[e]], 1.0f);
}
__global__ void deg_rsqrt_kernel(float* deg, int n) {
    int i = blockIdx.x * blockDim.x + threadIdx.x;
    if (i < n) deg[i] = rsqrtf(deg[i]);             // deg >= 1 (self-loop)
}

// agg[i][:] = m[i][:] * dinv[i]^2   (self-loop contribution, also zero-init)
__global__ void agg_init_kernel(const float* __restrict__ m, const float* __restrict__ dinv,
                                float* __restrict__ agg, int N) {
    int i = blockIdx.x * blockDim.x + threadIdx.x;  // one float4 per thread
    if (i >= N * 32) return;
    int node = i >> 5;
    int off  = (i & 31) * 4;
    float s = dinv[node]; s *= s;
    const float4 v = *(const float4*)&m[(size_t)node * 128 + off];
    float4 o; o.x = v.x * s; o.y = v.y * s; o.z = v.z * s; o.w = v.w * s;
    *(float4*)&agg[(size_t)node * 128 + off] = o;
}

// One wave32 per edge: lane handles 4 contiguous channels.
// m rows (51 MB) live in the 192 MB L2 -> gathers + f32 atomics stay on-die.
__global__ __launch_bounds__(256)
void scatter_edges_kernel(const int* __restrict__ eidx, const float* __restrict__ dinv,
                          const float* __restrict__ m, float* __restrict__ agg, int E) {
    int gw = (int)((blockIdx.x * blockDim.x + threadIdx.x) >> 5);
    int lane = threadIdx.x & 31;
    if (gw >= E) return;
    int row = eidx[gw];          // source j
    int col = eidx[E + gw];      // target i
    float nrm = dinv[row] * dinv[col];
    const float4 v = *(const float4*)&m[(size_t)row * 128 + lane * 4];
    float* dst = &agg[(size_t)col * 128 + lane * 4];
    atomicAdd(dst + 0, v.x * nrm);
    atomicAdd(dst + 1, v.y * nrm);
    atomicAdd(dst + 2, v.z * nrm);
    atomicAdd(dst + 3, v.w * nrm);
}

// h[i][c] = relu(agg[i][c] + b[c])  (in place: agg aliases h)
__global__ void bias_relu_kernel(float* __restrict__ h, const float* __restrict__ b, int N) {
    int i = blockIdx.x * blockDim.x + threadIdx.x;
    if (i >= N * 32) return;
    int node = i >> 5;
    int off  = (i & 31) * 4;
    float4 v = *(const float4*)&h[(size_t)node * 128 + off];
    v.x = fmaxf(v.x + b[off + 0], 0.0f);
    v.y = fmaxf(v.y + b[off + 1], 0.0f);
    v.z = fmaxf(v.z + b[off + 2], 0.0f);
    v.w = fmaxf(v.w + b[off + 3], 0.0f);
    *(float4*)&h[(size_t)node * 128 + off] = v;
}

__global__ void zero_kernel(float* p, int n) {
    int i = blockIdx.x * blockDim.x + threadIdx.x;
    if (i < n) p[i] = 0.0f;
}

__global__ void pool_kernel(const float* __restrict__ h, const int* __restrict__ batch,
                            float* __restrict__ g, int N) {
    int i = blockIdx.x * blockDim.x + threadIdx.x;
    if (i >= N * 32) return;
    int node = i >> 5;
    int off  = (i & 31) * 4;
    int gi = batch[node];
    const float4 v = *(const float4*)&h[(size_t)node * 128 + off];
    float* dst = &g[(size_t)gi * 128 + off];
    atomicAdd(dst + 0, v.x);
    atomicAdd(dst + 1, v.y);
    atomicAdd(dst + 2, v.z);
    atomicAdd(dst + 3, v.w);
}

__global__ void decode_kernel(const float* __restrict__ g, const float* __restrict__ W,
                              const float* __restrict__ b, float* __restrict__ out,
                              int G, int OUTD) {
    int t = blockIdx.x * blockDim.x + threadIdx.x;
    if (t >= G * OUTD) return;
    int gi = t / OUTD, o = t % OUTD;
    float s = b[o];
#pragma unroll 8
    for (int k = 0; k < 128; ++k) s += g[gi * 128 + k] * W[k * OUTD + o];
    out[t] = s;
}

// ---------------------------------------------------------------------------
extern "C" void kernel_launch(void* const* d_in, const int* in_sizes, int n_in,
                              void* d_out, int out_size, void* d_ws, size_t ws_size,
                              hipStream_t stream) {
    const float* x      = (const float*)d_in[0];
    const float* enc_W  = (const float*)d_in[1];
    const float* enc_b  = (const float*)d_in[2];
    const float* conv_W = (const float*)d_in[3];
    const float* conv_b = (const float*)d_in[4];
    const float* dec_W  = (const float*)d_in[5];
    const float* dec_b  = (const float*)d_in[6];
    const int*   eidx   = (const int*)d_in[7];   // [2, E] flat: row then col
    const int*   batch  = (const int*)d_in[8];

    const int N    = in_sizes[0] / HID;
    const int E    = in_sizes[7] / 2;
    const int L    = in_sizes[3] / (HID * HID);
    const int OUTD = in_sizes[6];
    const int G    = out_size / OUTD;

    // workspace layout (agg aliases h: h is dead once m = h@W is computed)
    float* h    = (float*)d_ws;                 // N*128  (also agg)
    float* m    = h + (size_t)N * HID;          // N*128
    float* dinv = m + (size_t)N * HID;          // N
    float* g    = dinv + N;                     // G*128

    const int T = 256;
    // GCN symmetric norm with self-loops
    deg_init_kernel <<<(N + T - 1) / T, T, 0, stream>>>(dinv, N);
    deg_count_kernel<<<(E + T - 1) / T, T, 0, stream>>>(eidx + E, dinv, E);
    deg_rsqrt_kernel<<<(N + T - 1) / T, T, 0, stream>>>(dinv, N);

    // encoder: h = relu(x @ enc_W + enc_b)
    gemm128_wmma_kernel<<<(N + 63) / 64, 128, 0, stream>>>(x, enc_W, enc_b, h, N, 1);

    for (int l = 0; l < L; ++l) {
        // m = h @ conv_W[l]
        gemm128_wmma_kernel<<<(N + 63) / 64, 128, 0, stream>>>(
            h, conv_W + (size_t)l * HID * HID, nullptr, m, N, 0);
        // agg (= h buffer) := self-loop term, then scatter-add edge messages
        agg_init_kernel<<<(N * 32 + T - 1) / T, T, 0, stream>>>(m, dinv, h, N);
        scatter_edges_kernel<<<(E + 7) / 8, T, 0, stream>>>(eidx, dinv, m, h, E);
        // h = relu(agg + conv_b[l])   (in place)
        bias_relu_kernel<<<(N * 32 + T - 1) / T, T, 0, stream>>>(h, conv_b + l * HID, N);
    }

    // global add pool + decoder
    zero_kernel  <<<(G * HID + T - 1) / T, T, 0, stream>>>(g, G * HID);
    pool_kernel  <<<(N * 32 + T - 1) / T, T, 0, stream>>>(h, batch, g, N);
    decode_kernel<<<(G * OUTD + T - 1) / T, T, 0, stream>>>(g, dec_W, dec_b,
                                                            (float*)d_out, G, OUTD);
}
